// PointActorTFN_52295521796465
// MI455X (gfx1250) — matrix operand order repaired and checked by reading, not compiled, not run on hardware
//
#include <hip/hip_runtime.h>
#include <hip/hip_fp16.h>
#include <hip/hip_bf16.h>

// ---------------- problem constants ----------------
constexpr int B_   = 64;
constexpr int N_   = 8192;
constexpr int NT_  = 4096;
constexpr int FIN_ = 6;
constexpr int H_   = 128;
constexpr float SCALE_ = 50.0f;

constexpr int NPTS = B_ * NT_;          // 262144 points that reach the output

// ---------------- WMMA tiling ----------------
typedef __attribute__((ext_vector_type(16))) _Float16 v16h;
typedef __attribute__((ext_vector_type(8)))  _Float16 v8h;
typedef __attribute__((ext_vector_type(8)))  float    v8f;

constexpr int WAVES   = 8;               // 8 wave32 per block
constexpr int TPB     = WAVES * 32;      // 256 threads
constexpr int TILES_W = 4;               // four 16-row tiles per wave
constexpr int ROWS_W  = TILES_W * 16;    // 64 points per wave
constexpr int PTS_BLK = WAVES * ROWS_W;  // 512 points per block-tile
constexpr int NTILES  = NPTS / PTS_BLK;  // 512

constexpr int WSTRIDE = 136;             // padded f16 stride (272B -> bank rotate)
constexpr int ASTRIDE = 136;

// dynamic LDS layout
constexpr int WT_HALFS  = 4 * H_ * WSTRIDE;            // 4x 128x128 f16 (transposed)
constexpr int ACT_HALFS = WAVES * ROWS_W * ASTRIDE;    // 512 x 136 halves (139,264 B)
constexpr int FLT_COUNT = FIN_*H_ + H_ + 4*H_ + 3*H_ + 4;
constexpr size_t LDS_BYTES = (size_t)(WT_HALFS + ACT_HALFS) * 2 + (size_t)FLT_COUNT * 4;
static_assert(LDS_BYTES <= 320*1024, "over WGP LDS");
static_assert((size_t)ACT_HALFS * 2 >= (size_t)H_ * H_ * 4, "stage buffer must fit in act");

// ---------------- TDM weight staging (gfx1250 Tensor Data Mover) ----------------
typedef __attribute__((ext_vector_type(4))) unsigned int tdm_v4u;
typedef __attribute__((ext_vector_type(4))) int          tdm_v4i;
typedef __attribute__((ext_vector_type(8))) int          tdm_v8i;

#if __has_builtin(__builtin_amdgcn_tensor_load_to_lds) && __has_builtin(__builtin_amdgcn_s_wait_tensorcnt)
#define USE_TDM 1
#pragma message("CDNA5: TDM tensor_load_to_lds path ENABLED")
#else
#define USE_TDM 0
#pragma message("CDNA5: TDM builtin not found - plain-load fallback")
#endif

#if USE_TDM
// DMA a 128x128 fp32 tile (contiguous, stride 128) from global into LDS at byte
// offset lds_off.  D# fields per CDNA5 ISA ch. 8 (group0: count/lds/global/type,
// group1: data_size=4B, tensor 128x128, tile 128x128x1, dim0 stride 128).
__device__ __forceinline__ void tdm_load_w128(const float* __restrict__ gsrc,
                                              unsigned lds_off)
{
    const unsigned long long ga = (unsigned long long)(uintptr_t)gsrc;
    tdm_v4u g0;
    g0[0] = 1u;                                         // count=1, user desc
    g0[1] = lds_off;                                    // lds_addr (bytes)
    g0[2] = (unsigned)(ga & 0xFFFFFFFFull);             // global_addr[31:0]
    g0[3] = (unsigned)((ga >> 32) & 0x01FFFFFFull)      // global_addr[56:32]
          | (2u << 30);                                 // type=2 ("image")
    tdm_v8i g1;
    g1[0] = (int)(2u << 16);        // workgroup_mask=0, data_size=2 (4 bytes)
    g1[1] = (int)(128u << 16);      // tensor_dim0[15:0]=128 (bits 79:48 span)
    g1[2] = (int)(128u << 16);      // tensor_dim0 hi=0 | tensor_dim1[15:0]=128
    g1[3] = (int)(128u << 16);      // tensor_dim1 hi=0 | tile_dim0=128
    g1[4] = (int)(128u);            // tile_dim1=128 | tile_dim2=0
    g1[5] = (int)128;               // tensor_dim0_stride[31:0]=128 (units of 4B)
    g1[6] = 0;                      // stride0 hi | stride1 lo
    g1[7] = 0;                      // stride1 hi
    const tdm_v4i z4 = {0, 0, 0, 0};
#if defined(__clang_major__) && (__clang_major__ >= 23)
    const tdm_v8i z8 = {0, 0, 0, 0, 0, 0, 0, 0};
    __builtin_amdgcn_tensor_load_to_lds(g0, g1, z4, z4, z8, 0);
#else
    __builtin_amdgcn_tensor_load_to_lds(g0, g1, z4, z4, 0);
#endif
}
#endif

// =====================================================================
// Kernel 1: fused 6-layer MLP, WMMA f32<=f16 for the four 128x128 layers
// =====================================================================
__global__ void __launch_bounds__(TPB, 1)
mlp_flow_kernel(const float* __restrict__ feat,
                const float* __restrict__ W1,  const float* __restrict__ b1,
                const float* __restrict__ W2,  const float* __restrict__ b2,
                const float* __restrict__ W3,  const float* __restrict__ b3,
                const float* __restrict__ Wm1, const float* __restrict__ bm1,
                const float* __restrict__ Wm2, const float* __restrict__ bm2,
                const float* __restrict__ Wm3, const float* __restrict__ bm3,
                float* __restrict__ flow_out)
{
    extern __shared__ char smem[];
    _Float16* wt  = (_Float16*)smem;                 // [4][H][WSTRIDE]  (WT[n][k])
    _Float16* act = wt + WT_HALFS;                   // [WAVES*64][ASTRIDE]
    float*    w1s  = (float*)(act + ACT_HALFS);      // [FIN][H]
    float*    b1s  = w1s  + FIN_*H_;                 // [H]
    float*    bLs  = b1s  + H_;                      // [4][H]
    float*    wm3s = bLs  + 4*H_;                    // [H][3]
    float*    bm3s = wm3s + H_*3;                    // [3]+pad

    const int tid  = threadIdx.x;
    const int wave = tid >> 5;
    const int lane = tid & 31;

    // ---- stage weights: fp32 global -> transposed fp16 LDS ----
    {
        const float* Wsrc[4] = {W2, W3, Wm1, Wm2};
#if USE_TDM
        // staging buffer aliases the (still dead) activation region
        float* stage = (float*)act;
        const unsigned stage_off = (unsigned)(WT_HALFS * 2);  // dyn LDS starts at 0
        #pragma unroll 1
        for (int l = 0; l < 4; ++l) {
            __syncthreads();                 // stage free (prev layer converted)
            if (wave == 0) {
                tdm_load_w128(Wsrc[l], stage_off);
                __builtin_amdgcn_s_wait_tensorcnt(0);
            }
            __syncthreads();                 // stage holds W[l] (fp32, row-major)
            _Float16* dst = wt + l * H_ * WSTRIDE;
            for (int idx = tid; idx < H_*H_; idx += TPB) {
                const int k = idx >> 7;              // input dim
                const int n = idx & (H_-1);          // output dim
                dst[n * WSTRIDE + k] = (_Float16)stage[idx];
            }
        }
        __syncthreads();
#else
        #pragma unroll 1
        for (int l = 0; l < 4; ++l) {
            const float* W = Wsrc[l];
            _Float16* dst = wt + l * H_ * WSTRIDE;
            for (int idx = tid; idx < H_*H_; idx += TPB) {
                const int k = idx >> 7;
                const int n = idx & (H_-1);
                dst[n * WSTRIDE + k] = (_Float16)W[idx];
            }
        }
#endif
        const float* bsrc[4] = {b2, b3, bm1, bm2};
        #pragma unroll 1
        for (int l = 0; l < 4; ++l)
            for (int i = tid; i < H_; i += TPB) bLs[l*H_ + i] = bsrc[l][i];
        for (int i = tid; i < FIN_*H_; i += TPB) w1s[i] = W1[i];
        for (int i = tid; i < H_;      i += TPB) b1s[i] = b1[i];
        for (int i = tid; i < H_*3;    i += TPB) wm3s[i] = Wm3[i];
        if (tid < 3) bm3s[tid] = bm3[tid];
        if (tid == 3) bm3s[3] = 0.f;
    }
    __syncthreads();

    const int lhalf = lane >> 4;        // 0: lanes 0-15, 1: lanes 16-31
    const int lrow  = lane & 15;
    const int wrow0 = wave * ROWS_W;    // first of this wave's 64 act rows

    #pragma unroll 1
    for (int tile = blockIdx.x; tile < NTILES; tile += gridDim.x) {
        const int p0 = tile * PTS_BLK;

        // ---------- layer 1: FIN(6) -> H(128), fp32 VALU, 2 points per lane ----------
        #pragma unroll
        for (int pt = 0; pt < 2; ++pt) {
            const int r   = lane + 32*pt;              // row within wave block
            const int gp  = p0 + wave*ROWS_W + r;
            const int bi  = gp >> 12;                  // / NT (4096)
            const int ni  = gp & (NT_-1);
            float f[FIN_];
            const float* fp = feat + ((size_t)bi * N_ + ni) * FIN_;
            #pragma unroll
            for (int j = 0; j < FIN_; ++j) f[j] = fp[j];
            _Float16* arow = act + (wrow0 + r) * ASTRIDE;
            #pragma unroll 4
            for (int h = 0; h < H_; ++h) {
                float acc = b1s[h];
                #pragma unroll
                for (int j = 0; j < FIN_; ++j)
                    acc = fmaf(f[j], w1s[j*H_ + h], acc);
                arow[h] = (_Float16)fmaxf(acc, 0.f);
            }
        }

        // ---------- layers 2..5: H -> H via v_wmma_f32_16x16x32_f16 ----------
        #pragma unroll 1
        for (int layer = 0; layer < 4; ++layer) {
            const _Float16* WT = wt + layer * H_ * WSTRIDE;
            const float*    bL = bLs + layer * H_;

            // A fragments for the four 16-row tiles (ISA 16-bit A layout)
            v16h a[TILES_W][4];
            {
                const int ka = lhalf * 8;
                #pragma unroll
                for (int t = 0; t < TILES_W; ++t) {
                    const _Float16* ar = act + (wrow0 + t*16 + lrow) * ASTRIDE;
                    #pragma unroll
                    for (int kt = 0; kt < 4; ++kt) {
                        const v8h lo = *(const v8h*)(ar + kt*32 +      ka);
                        const v8h hi = *(const v8h*)(ar + kt*32 + 16 + ka);
                        a[t][kt] = __builtin_shufflevector(lo, hi,
                                   0,1,2,3,4,5,6,7,8,9,10,11,12,13,14,15);
                    }
                }
            }

            #pragma unroll
            for (int nt = 0; nt < 8; ++nt) {
                const int col = nt*16 + lrow;       // output column
                // batch-load all 4 B fragments for this column tile
                v16h bf[4];
                #pragma unroll
                for (int kt = 0; kt < 4; ++kt) {
                    const _Float16* wp = WT + col * WSTRIDE + kt*32 + lhalf*8;
                    const v8h blo = *(const v8h*)wp;
                    const v8h bhi = *(const v8h*)(wp + 16);
                    bf[kt] = __builtin_shufflevector(blo, bhi,
                             0,1,2,3,4,5,6,7,8,9,10,11,12,13,14,15);
                }
                const float bv = bL[col];
                v8f c[TILES_W];
                #pragma unroll
                for (int t = 0; t < TILES_W; ++t)
                    #pragma unroll
                    for (int i = 0; i < 8; ++i) c[t][i] = bv;

                // 4 independent accumulation chains, interleaved per K-step
                #pragma unroll
                for (int kt = 0; kt < 4; ++kt)
                    #pragma unroll
                    for (int t = 0; t < TILES_W; ++t)
                        c[t] = __builtin_amdgcn_wmma_f32_16x16x32_f16(
                                false, a[t][kt], false, bf[kt],
                                (short)0, c[t], false, false);

                // relu + fp16 writeback in place (A already in regs, rows wave-private)
                #pragma unroll
                for (int t = 0; t < TILES_W; ++t)
                    #pragma unroll
                    for (int v = 0; v < 8; ++v) {
                        const int r = wrow0 + t*16 + v + lhalf*8;
                        act[r * ASTRIDE + col] = (_Float16)fmaxf(c[t][v], 0.f);
                    }
            }
        }

        // ---------- layer 6: H -> 3, fp32 VALU, 2 points per lane ----------
        #pragma unroll
        for (int pt = 0; pt < 2; ++pt) {
            const int r  = lane + 32*pt;
            const int gp = p0 + wave*ROWS_W + r;
            const _Float16* ar = act + (wrow0 + r) * ASTRIDE;
            float acc0 = bm3s[0], acc1 = bm3s[1], acc2 = bm3s[2];
            #pragma unroll 8
            for (int k = 0; k < H_; ++k) {
                const float av = (float)ar[k];
                acc0 = fmaf(av, wm3s[k*3 + 0], acc0);
                acc1 = fmaf(av, wm3s[k*3 + 1], acc1);
                acc2 = fmaf(av, wm3s[k*3 + 2], acc2);
            }
            float* fo = flow_out + (size_t)gp * 3;
            fo[0] = acc0; fo[1] = acc1; fo[2] = acc2;
        }
    }
}

// =====================================================================
// Kernel 2: per-batch weighted-Kabsch rigid fit (one block per batch)
// =====================================================================
__device__ inline float det3(const float* m) {
    return m[0]*(m[4]*m[8]-m[5]*m[7])
         - m[1]*(m[3]*m[8]-m[5]*m[6])
         + m[2]*(m[3]*m[7]-m[4]*m[6]);
}

__global__ void __launch_bounds__(256)
kabsch_kernel(const float* __restrict__ pos,
              const float* __restrict__ flow,
              float* __restrict__ out)
{
    __shared__ float red[16];       // Sx(3), Sy(3), Sxy(9)
    __shared__ float Rs[9], cxs[3], cys[3];

    const int b = blockIdx.x;
    const int tid = threadIdx.x;
    if (tid < 16) red[tid] = 0.f;
    __syncthreads();

    float sx[3] = {0,0,0}, sy[3] = {0,0,0}, sxy[9] = {0,0,0,0,0,0,0,0,0};
    for (int n = tid; n < NT_; n += blockDim.x) {
        const float* pp = pos  + ((size_t)b * N_  + n) * 3;
        const float* fp = flow + ((size_t)b * NT_ + n) * 3;
        float x[3], y[3];
        #pragma unroll
        for (int i = 0; i < 3; ++i) {
            x[i] = pp[i] * SCALE_;
            y[i] = x[i] + fp[i];
            sx[i] += x[i]; sy[i] += y[i];
        }
        #pragma unroll
        for (int i = 0; i < 3; ++i)
            #pragma unroll
            for (int j = 0; j < 3; ++j)
                sxy[i*3+j] = fmaf(x[i], y[j], sxy[i*3+j]);
    }
    #pragma unroll
    for (int i = 0; i < 3; ++i) { atomicAdd(&red[i], sx[i]); atomicAdd(&red[3+i], sy[i]); }
    #pragma unroll
    for (int i = 0; i < 9; ++i) atomicAdd(&red[6+i], sxy[i]);
    __syncthreads();

    if (tid == 0) {
        const float inv = 1.0f / (float)NT_;
        float cx[3], cy[3];
        for (int i = 0; i < 3; ++i) { cx[i] = red[i]*inv; cy[i] = red[3+i]*inv; }
        float Hm[9];
        for (int i = 0; i < 3; ++i)
            for (int j = 0; j < 3; ++j)
                Hm[i*3+j] = red[6+i*3+j] - (float)NT_ * cx[i]*cy[j];

        // A = Hm^T Hm (symmetric PSD), Jacobi eigen-decomposition -> V, lam
        float A[9];
        for (int i = 0; i < 3; ++i)
            for (int j = 0; j < 3; ++j)
                A[i*3+j] = Hm[0*3+i]*Hm[0*3+j] + Hm[1*3+i]*Hm[1*3+j] + Hm[2*3+i]*Hm[2*3+j];
        float V[9] = {1,0,0, 0,1,0, 0,0,1};
        const int PQ[3][2] = {{0,1},{0,2},{1,2}};
        for (int sweep = 0; sweep < 16; ++sweep) {
            for (int r = 0; r < 3; ++r) {
                const int p = PQ[r][0], q = PQ[r][1];
                const float apq = A[p*3+q];
                if (fabsf(apq) < 1e-24f) continue;
                const float theta = (A[q*3+q] - A[p*3+p]) / (2.0f * apq);
                const float t = (theta >= 0.f ? 1.f : -1.f) /
                                (fabsf(theta) + sqrtf(theta*theta + 1.f));
                const float c = 1.0f / sqrtf(t*t + 1.f);
                const float s = t * c;
                for (int k = 0; k < 3; ++k) {
                    const float akp = A[k*3+p], akq = A[k*3+q];
                    A[k*3+p] = c*akp - s*akq;  A[k*3+q] = s*akp + c*akq;
                }
                for (int k = 0; k < 3; ++k) {
                    const float apk = A[p*3+k], aqk = A[q*3+k];
                    A[p*3+k] = c*apk - s*aqk;  A[q*3+k] = s*apk + c*aqk;
                }
                for (int k = 0; k < 3; ++k) {
                    const float vkp = V[k*3+p], vkq = V[k*3+q];
                    V[k*3+p] = c*vkp - s*vkq;  V[k*3+q] = s*vkp + c*vkq;
                }
            }
        }
        float lam[3] = {A[0], A[4], A[8]};
        for (int i = 0; i < 2; ++i)
            for (int j = i+1; j < 3; ++j)
                if (lam[j] > lam[i]) {
                    float tl = lam[i]; lam[i] = lam[j]; lam[j] = tl;
                    for (int k = 0; k < 3; ++k) {
                        float tv = V[k*3+i]; V[k*3+i] = V[k*3+j]; V[k*3+j] = tv;
                    }
                }
        float U[9];
        for (int j = 0; j < 3; ++j) {
            const float sj = sqrtf(fmaxf(lam[j], 0.f));
            if (sj > 1e-8f || j < 2) {
                const float invs = 1.0f / fmaxf(sj, 1e-20f);
                for (int i = 0; i < 3; ++i)
                    U[i*3+j] = (Hm[i*3+0]*V[0*3+j] + Hm[i*3+1]*V[1*3+j] + Hm[i*3+2]*V[2*3+j]) * invs;
            } else {
                U[0*3+2] = U[1*3+0]*U[2*3+1] - U[2*3+0]*U[1*3+1];
                U[1*3+2] = U[2*3+0]*U[0*3+1] - U[0*3+0]*U[2*3+1];
                U[2*3+2] = U[0*3+0]*U[1*3+1] - U[1*3+0]*U[0*3+1];
            }
        }
        // d = det(V U^T);  R = V diag(1,1,d) U^T
        float P[9];
        for (int i = 0; i < 3; ++i)
            for (int k = 0; k < 3; ++k)
                P[i*3+k] = V[i*3+0]*U[k*3+0] + V[i*3+1]*U[k*3+1] + V[i*3+2]*U[k*3+2];
        const float d = det3(P);
        for (int i = 0; i < 3; ++i)
            for (int k = 0; k < 3; ++k)
                Rs[i*3+k] = V[i*3+0]*U[k*3+0] + V[i*3+1]*U[k*3+1] + d * V[i*3+2]*U[k*3+2];
        for (int i = 0; i < 3; ++i) { cxs[i] = cx[i]; cys[i] = cy[i]; }
    }
    __syncthreads();

    // batch_flows = (Xc @ R^T + cy) - xyz
    for (int n = tid; n < NT_; n += blockDim.x) {
        const float* pp = pos + ((size_t)b * N_ + n) * 3;
        float x[3], Xc[3];
        #pragma unroll
        for (int i = 0; i < 3; ++i) { x[i] = pp[i] * SCALE_; Xc[i] = x[i] - cxs[i]; }
        float* op = out + ((size_t)b * NT_ + n) * 3;
        #pragma unroll
        for (int j = 0; j < 3; ++j) {
            const float t = cys[j] + Xc[0]*Rs[j*3+0] + Xc[1]*Rs[j*3+1] + Xc[2]*Rs[j*3+2];
            op[j] = t - x[j];
        }
    }
}

// =====================================================================
// host launcher
// =====================================================================
extern "C" void kernel_launch(void* const* d_in, const int* in_sizes, int n_in,
                              void* d_out, int out_size, void* d_ws, size_t ws_size,
                              hipStream_t stream)
{
    (void)in_sizes; (void)n_in; (void)out_size; (void)d_ws; (void)ws_size;
    const float* pos  = (const float*)d_in[0];
    const float* feat = (const float*)d_in[1];
    const float* W1   = (const float*)d_in[2];
    const float* b1   = (const float*)d_in[3];
    const float* W2   = (const float*)d_in[4];
    const float* b2   = (const float*)d_in[5];
    const float* W3   = (const float*)d_in[6];
    const float* b3   = (const float*)d_in[7];
    const float* Wm1  = (const float*)d_in[8];
    const float* bm1  = (const float*)d_in[9];
    const float* Wm2  = (const float*)d_in[10];
    const float* bm2  = (const float*)d_in[11];
    const float* Wm3  = (const float*)d_in[12];
    const float* bm3  = (const float*)d_in[13];

    float* out_batch_flows = (float*)d_out;                      // [B, NT, 3]
    float* out_flow        = out_batch_flows + (size_t)B_*NT_*3; // [B, NT, 3]

    // allow >64KB dynamic LDS (CDNA5 WGP has 320KB)
    (void)hipFuncSetAttribute(reinterpret_cast<const void*>(mlp_flow_kernel),
                              hipFuncAttributeMaxDynamicSharedMemorySize,
                              (int)LDS_BYTES);

    mlp_flow_kernel<<<NTILES, TPB, LDS_BYTES, stream>>>(
        feat, W1, b1, W2, b2, W3, b3, Wm1, bm1, Wm2, bm2, Wm3, bm3, out_flow);

    kabsch_kernel<<<B_, 256, 0, stream>>>(pos, out_flow, out_batch_flows);
}
